// MultichannelMultiheadAttention_11879879544484
// MI455X (gfx1250) — matrix operand order repaired and checked by reading, not compiled, not run on hardware
//
#include <hip/hip_runtime.h>
#include <hip/hip_bf16.h>
#include <math.h>

// MI455X / gfx1250, wave32. WMMA bf16 16x16x32, f32 accumulate.

typedef __attribute__((ext_vector_type(16))) __bf16 v16bf;
typedef __attribute__((ext_vector_type(8)))  float  v8f;

#define B_     2
#define C_     8
#define F_     256
#define W_     1024
#define HD_    32
#define GP     40   // LDS row pitch in bf16 elems (pad 32 -> 40, keeps 16B alignment)

__device__ __forceinline__ __bf16 f2bf(float f) {
  unsigned u = __builtin_bit_cast(unsigned, f);
  unsigned r = u + 0x7FFFu + ((u >> 16) & 1u);   // round-to-nearest-even
  unsigned short h = (unsigned short)(r >> 16);
  return __builtin_bit_cast(__bf16, h);
}

__device__ __forceinline__ v8f zero8() {
  v8f v;
  for (int i = 0; i < 8; ++i) v[i] = 0.0f;
  return v;
}

__device__ __forceinline__ v8f wmma_bf16(v16bf a, v16bf b, v8f c) {
  // (neg_a, A, neg_b, B, c_mod, C, reuse_a, reuse_b)
  return __builtin_amdgcn_wmma_f32_16x16x32_bf16(false, a, false, b, (short)0, c,
                                                 false, false);
}

union BF16Frag {
  __bf16 h[16];
  uint4  q[2];
  v16bf  v;
};

// ---------------------------------------------------------------------------
// Per-(b,c) GEMM: Y[g,w] = sum_h W[c][g,h] * X[b,c][h,w]
// M=256 (g), K=256 (h), N=1024 (w). Tile 128x128 per block, 8 waves.
// ---------------------------------------------------------------------------
__global__ __launch_bounds__(256)
void mcl_gemm_k(const float* __restrict__ Wt, const float* __restrict__ X,
                float* __restrict__ Y) {
  __shared__ __bf16 ldsA[128 * GP];   // [m][k]
  __shared__ __bf16 ldsB[128 * GP];   // [n][k] (transposed X tile)
  const int bc = blockIdx.z;
  const int c  = bc & (C_ - 1);
  const int m_base = blockIdx.y * 128;
  const int n_base = blockIdx.x * 128;
  const float* Wp = Wt + (size_t)c * F_ * F_;
  const float* Xp = X + (size_t)bc * F_ * W_;
  float*       Yp = Y + (size_t)bc * F_ * W_;

  const int tid  = threadIdx.x;
  const int wave = tid >> 5;
  const int lane = tid & 31;
  const int half = (lane >> 4) & 1;
  const int l16  = lane & 15;
  const int wm = (wave >> 2) * 64;   // wave M offset (0/64)
  const int wn = (wave & 3) * 32;    // wave N offset (0/32/64/96)

  v8f acc[4][2];
  for (int i = 0; i < 4; ++i)
    for (int j = 0; j < 2; ++j) acc[i][j] = zero8();

  for (int k0 = 0; k0 < F_; k0 += 32) {
    __syncthreads();
    // A tile: 128 x 32 f32 -> bf16 LDS (row-major)
    for (int i = 0; i < 4; ++i) {
      int idx = tid + i * 256;           // 0..1023
      int row = idx >> 3;                // 0..127
      int col = (idx & 7) << 2;          // 0..28
      float4 f = *(const float4*)(Wp + (size_t)(m_base + row) * F_ + k0 + col);
      __bf16* d = &ldsA[row * GP + col];
      d[0] = f2bf(f.x); d[1] = f2bf(f.y); d[2] = f2bf(f.z); d[3] = f2bf(f.w);
      if (k0 + 32 < F_)
        __builtin_prefetch(Wp + (size_t)(m_base + row) * F_ + k0 + 32 + col, 0, 0);
    }
    // B tile: 32 x 128 f32, stored transposed as [n][k]
    for (int i = 0; i < 4; ++i) {
      int idx = tid + i * 256;
      int kr  = idx >> 5;                // 0..31
      int col = (idx & 31) << 2;         // 0..124
      float4 f = *(const float4*)(Xp + (size_t)(k0 + kr) * W_ + n_base + col);
      ldsB[(col + 0) * GP + kr] = f2bf(f.x);
      ldsB[(col + 1) * GP + kr] = f2bf(f.y);
      ldsB[(col + 2) * GP + kr] = f2bf(f.z);
      ldsB[(col + 3) * GP + kr] = f2bf(f.w);
    }
    __syncthreads();

    BF16Frag afr[4], bfr[2];
    // A fragment: lane row = l16; half 0 holds K {0..7,16..23}, half 1 {8..15,24..31}
    for (int mi = 0; mi < 4; ++mi) {
      const __bf16* p = &ldsA[(wm + mi * 16 + l16) * GP + (half ? 8 : 0)];
      afr[mi].q[0] = *(const uint4*)(p);
      afr[mi].q[1] = *(const uint4*)(p + 16);
    }
    // B fragment: lane col = l16; half 0 holds K {0..7,16..23}, half 1 {8..15,24..31}
    for (int ni = 0; ni < 2; ++ni) {
      const __bf16* p = &ldsB[(wn + ni * 16 + l16) * GP + (half ? 8 : 0)];
      bfr[ni].q[0] = *(const uint4*)(p);
      bfr[ni].q[1] = *(const uint4*)(p + 16);
    }
    for (int mi = 0; mi < 4; ++mi)
      for (int ni = 0; ni < 2; ++ni)
        acc[mi][ni] = wmma_bf16(afr[mi].v, bfr[ni].v, acc[mi][ni]);
  }

  // C layout: VGPR r -> row r (+8 for upper half-wave), lane l16 -> col
  for (int mi = 0; mi < 4; ++mi)
    for (int ni = 0; ni < 2; ++ni) {
      int gm = m_base + wm + mi * 16 + half * 8;
      int gn = n_base + wn + ni * 16 + l16;
      for (int r = 0; r < 8; ++r)
        Yp[(size_t)(gm + r) * W_ + gn] = acc[mi][ni][r];
    }
}

// ---------------------------------------------------------------------------
// conv1x3 over w with channel mix (c=8), zero-padded
// ---------------------------------------------------------------------------
__global__ __launch_bounds__(256)
void conv1x3_k(const float* __restrict__ in, const float* __restrict__ wgt,
               float* __restrict__ out) {
  __shared__ float ws[C_ * C_ * 3];
  if (threadIdx.x < C_ * C_ * 3) ws[threadIdx.x] = wgt[threadIdx.x];
  __syncthreads();
  int idx = blockIdx.x * blockDim.x + threadIdx.x;   // b,co,h,p
  int p  = idx & (W_ - 1);
  int h  = (idx >> 10) & (F_ - 1);
  int co = (idx >> 18) & (C_ - 1);
  int b  = idx >> 21;
  float acc = 0.0f;
  for (int ci = 0; ci < C_; ++ci) {
    const float* row = in + ((size_t)(b * C_ + ci) * F_ + h) * W_;
    float xl = (p > 0)      ? row[p - 1] : 0.0f;
    float xc = row[p];
    float xr = (p < W_ - 1) ? row[p + 1] : 0.0f;
    const float* wr = &ws[(co * C_ + ci) * 3];
    acc += wr[0] * xl + wr[1] * xc + wr[2] * xr;
  }
  out[idx] = acc;
}

// ---------------------------------------------------------------------------
// interleaved RoPE: seq = w, dim = hd = 32; one thread per (b,c,head,pair,p)
// ---------------------------------------------------------------------------
__global__ __launch_bounds__(256)
void rope_k(const float* __restrict__ in, float* __restrict__ out) {
  int idx = blockIdx.x * blockDim.x + threadIdx.x;   // bc,head,i,p  (2^21 total)
  int p    = idx & (W_ - 1);
  int i    = (idx >> 10) & 15;                       // pair index 0..15
  int head = (idx >> 14) & 7;
  int bc   = idx >> 17;
  size_t off = ((size_t)bc * F_ + head * HD_ + 2 * i) * W_ + p;
  float x0 = in[off];
  float x1 = in[off + W_];
  float fr = (float)p * powf(10000.0f, -(float)i * (1.0f / 16.0f));
  float sn, cs;
  __sincosf(fr, &sn, &cs);
  out[off]      = x0 * cs - x1 * sn;
  out[off + W_] = x1 * cs + x0 * sn;
}

// ---------------------------------------------------------------------------
// Flash attention per (b,c,head): q,k,v are [seq=1024, d=32] stored d-major
// (element (p,d) at base + d*1024 + p). One block = 4 waves = 64 q rows.
// ---------------------------------------------------------------------------
__global__ __launch_bounds__(128)
void attn_k(const float* __restrict__ q, const float* __restrict__ k,
            const float* __restrict__ v, float* __restrict__ o) {
  __shared__ __bf16 ldsP[4][16 * GP];   // per-wave P tile (16 rows x 32 keys)
  const int tid  = threadIdx.x;
  const int wave = tid >> 5;
  const int lane = tid & 31;
  const int half = (lane >> 4) & 1;
  const int l16  = lane & 15;
  const int q0   = blockIdx.x * 64 + wave * 16;
  const size_t base = (size_t)blockIdx.y * (HD_ * W_);   // g = bc*8+head

  // Q fragment (A layout): lane row = q0+l16, 16 consecutive d (half 1 -> d 16..31)
  BF16Frag qa;
  {
    int row = q0 + l16;
    for (int e = 0; e < 16; ++e) {
      int d = e + (half ? 16 : 0);
      qa.h[e] = f2bf(q[base + (size_t)d * W_ + row]);
    }
  }

  float m_s[8], l_s[8];
  for (int r = 0; r < 8; ++r) { m_s[r] = -1e30f; l_s[r] = 0.0f; }
  v8f o0 = zero8(), o1 = zero8();

  for (int kb = 0; kb < W_; kb += 32) {
    // K^T fragments (B layout): lane col = key, elements -> consecutive d
    BF16Frag kf0, kf1;
    {
      int key0 = kb + l16, key1 = kb + 16 + l16;
      int dbase = half ? 16 : 0;
      for (int e = 0; e < 16; ++e) {
        size_t doff = base + (size_t)(dbase + e) * W_;
        kf0.h[e] = f2bf(k[doff + key0]);
        kf1.h[e] = f2bf(k[doff + key1]);
      }
    }
    v8f s0 = wmma_bf16(qa.v, kf0.v, zero8());
    v8f s1 = wmma_bf16(qa.v, kf1.v, zero8());

    // online softmax; row r lives in one 16-lane half -> xor masks 1,2,4,8
    for (int r = 0; r < 8; ++r) {
      float a0 = s0[r] * 0.0625f;               // / sqrt(256)
      float a1 = s1[r] * 0.0625f;
      float mx = fmaxf(a0, a1);
      for (int msk = 1; msk <= 8; msk <<= 1) mx = fmaxf(mx, __shfl_xor(mx, msk, 32));
      float mnew  = fmaxf(m_s[r], mx);
      float scale = __expf(m_s[r] - mnew);
      float p0 = __expf(a0 - mnew);
      float p1 = __expf(a1 - mnew);
      float rs = p0 + p1;
      for (int msk = 1; msk <= 8; msk <<= 1) rs += __shfl_xor(rs, msk, 32);
      l_s[r] = l_s[r] * scale + rs;
      m_s[r] = mnew;
      o0[r] *= scale;
      o1[r] *= scale;
      __bf16* prow = &ldsP[wave][(r + half * 8) * GP];
      prow[l16]      = f2bf(p0);
      prow[16 + l16] = f2bf(p1);
    }

    // reload P as A fragment (C-layout -> A-layout via LDS, same wave, in-order)
    BF16Frag pa;
    {
      const __bf16* pp = &ldsP[wave][l16 * GP + (half ? 16 : 0)];
      pa.q[0] = *(const uint4*)(pp);
      pa.q[1] = *(const uint4*)(pp + 8);
    }
    // V fragments (B layout): lane col = d, elements -> 16 consecutive keys
    BF16Frag vf0, vf1;
    {
      int koff = kb + (half ? 16 : 0);
      const float* v0p = v + base + (size_t)l16 * W_ + koff;
      const float* v1p = v + base + (size_t)(16 + l16) * W_ + koff;
      for (int e = 0; e < 16; ++e) {
        vf0.h[e] = f2bf(v0p[e]);
        vf1.h[e] = f2bf(v1p[e]);
      }
    }
    o0 = wmma_bf16(pa.v, vf0.v, o0);
    o1 = wmma_bf16(pa.v, vf1.v, o1);
  }

  for (int r = 0; r < 8; ++r) {
    float inv = 1.0f / l_s[r];
    int row = q0 + r + half * 8;
    o[base + (size_t)l16 * W_ + row]        = o0[r] * inv;
    o[base + (size_t)(16 + l16) * W_ + row] = o1[r] * inv;
  }
}

// ---------------------------------------------------------------------------
// out[b,d,h,w] = sum_c a[b,c,h,w] * dw[d,c]
// ---------------------------------------------------------------------------
__global__ __launch_bounds__(256)
void chmix_k(const float* __restrict__ a, const float* __restrict__ dw,
             float* __restrict__ out) {
  __shared__ float ws[C_ * C_];
  if (threadIdx.x < C_ * C_) ws[threadIdx.x] = dw[threadIdx.x];
  __syncthreads();
  int idx = blockIdx.x * blockDim.x + threadIdx.x;   // b,h,p  (2^19 total)
  int p = idx & (W_ - 1);
  int h = (idx >> 10) & (F_ - 1);
  int b = idx >> 18;
  float xi[C_];
  for (int c = 0; c < C_; ++c)
    xi[c] = a[((size_t)(b * C_ + c) * F_ + h) * W_ + p];
  for (int d = 0; d < C_; ++d) {
    float s = 0.0f;
    for (int c = 0; c < C_; ++c) s += ws[d * C_ + c] * xi[c];
    out[((size_t)(b * C_ + d) * F_ + h) * W_ + p] = s;
  }
}

// ---------------------------------------------------------------------------
extern "C" void kernel_launch(void* const* d_in, const int* in_sizes, int n_in,
                              void* d_out, int out_size, void* d_ws, size_t ws_size,
                              hipStream_t stream) {
  (void)in_sizes; (void)n_in; (void)out_size; (void)ws_size;
  const float* x      = (const float*)d_in[0];
  const float* q_pw   = (const float*)d_in[1];
  const float* q_cw   = (const float*)d_in[2];
  const float* k_pw   = (const float*)d_in[3];
  const float* k_cw   = (const float*)d_in[4];
  const float* v_pw   = (const float*)d_in[5];
  const float* v_cw   = (const float*)d_in[6];
  const float* out_pw = (const float*)d_in[7];
  const float* out_dw = (const float*)d_in[8];
  float* out = (float*)d_out;

  const size_t TSZ = (size_t)B_ * C_ * F_ * W_;   // 4,194,304 elems (16 MB)
  float* ws0 = (float*)d_ws;
  float* ws1 = ws0 + TSZ;
  float* ws2 = ws1 + TSZ;                          // needs 48 MB of d_ws

  dim3 gGemm(W_ / 128, F_ / 128, B_ * C_);         // (8,2,16)
  const int nConvBlk = (int)(TSZ / 256);           // 16384
  const int nRopeBlk = (int)(TSZ / 2 / 256);       // 8192
  const int nMixBlk  = (int)(TSZ / C_ / 256);      // 2048
  dim3 gAttn(W_ / 64, B_ * C_ * (F_ / HD_));       // (16,128)

  // Q = rope(conv(mcl(x, q_pw), q_cw))
  mcl_gemm_k<<<gGemm, 256, 0, stream>>>(q_pw, x, ws0);
  conv1x3_k<<<nConvBlk, 256, 0, stream>>>(ws0, q_cw, ws1);
  rope_k<<<nRopeBlk, 256, 0, stream>>>(ws1, ws0);          // Q -> ws0
  // K
  mcl_gemm_k<<<gGemm, 256, 0, stream>>>(k_pw, x, ws1);
  conv1x3_k<<<nConvBlk, 256, 0, stream>>>(ws1, k_cw, ws2);
  rope_k<<<nRopeBlk, 256, 0, stream>>>(ws2, ws1);          // K -> ws1
  // V (use d_out as scratch; fully rewritten later)
  mcl_gemm_k<<<gGemm, 256, 0, stream>>>(v_pw, x, ws2);
  conv1x3_k<<<nConvBlk, 256, 0, stream>>>(ws2, v_cw, out); // V -> d_out
  // attention
  attn_k<<<gAttn, 128, 0, stream>>>(ws0, ws1, out, ws2);   // A -> ws2
  // output projection
  chmix_k<<<nMixBlk, 256, 0, stream>>>(ws2, out_dw, ws0);
  mcl_gemm_k<<<gGemm, 256, 0, stream>>>(out_pw, ws0, out);
}
